// PCA_NN_ADD_PCA_47278999995171
// MI455X (gfx1250) — matrix operand order repaired and checked by reading, not compiled, not run on hardware
//
#include <hip/hip_runtime.h>

typedef __bf16 bf16_t;
typedef __attribute__((ext_vector_type(16))) __bf16 v16bf;
typedef __attribute__((ext_vector_type(8)))  __bf16 v8bf;
typedef __attribute__((ext_vector_type(8)))  float  v8f;
typedef int v4i_vs __attribute__((vector_size(16)));   // int __vector(4), matches builtin param

union Frag { v16bf v; v8bf h[2]; };

#if __has_builtin(__builtin_amdgcn_global_load_async_to_lds_b128)
#define HAS_ASYNC_LDS 1
#else
#define HAS_ASYNC_LDS 0
#endif

__device__ __forceinline__ void g2l_async16(void* ldsp, const void* gp) {
#if HAS_ASYNC_LDS
  __builtin_amdgcn_global_load_async_to_lds_b128(
      (__attribute__((address_space(1))) v4i_vs*)(gp),
      (__attribute__((address_space(3))) v4i_vs*)(ldsp), 0, 0);
#else
  *(float4*)ldsp = *(const float4*)gp;
#endif
}

__device__ __forceinline__ void wait_async_lds() {
#if HAS_ASYNC_LDS
#if __has_builtin(__builtin_amdgcn_s_wait_asynccnt)
  __builtin_amdgcn_s_wait_asynccnt(0);
#else
  asm volatile("s_wait_asynccnt 0x0" ::: "memory");
#endif
#endif
}

__device__ __forceinline__ float leaky_f(float v) { return v > 0.f ? v : 0.1f * v; }

// ---------------- conversion kernels ----------------
__global__ __launch_bounds__(256) void cvt_bf16(const float* __restrict__ in,
                                                bf16_t* __restrict__ out, long n) {
  long i = (long)blockIdx.x * blockDim.x + threadIdx.x;
  if (i < n) out[i] = (bf16_t)in[i];
}

// transpose-convert: in [batch][K][N] f32 -> out [batch][N][K] bf16
__global__ __launch_bounds__(256) void cvt_t_bf16(const float* __restrict__ in,
                                                  bf16_t* __restrict__ out,
                                                  int K, int N, long total) {
  long i = (long)blockIdx.x * blockDim.x + threadIdx.x;
  if (i >= total) return;
  long kn = (long)K * N;
  long b = i / kn;
  long rem = i - b * kn;
  int k = (int)(rem / N);
  int n = (int)(rem - (long)k * N);
  out[b * kn + (long)n * K + k] = (bf16_t)in[i];
}

// ---------------- generic WMMA GEMM (double-buffered, async staging) ----------------
// C[M][N] = act( A[M][K] @ Wt[N][K]^T + bias[N] + *sbias )
// block: 256 threads = 8 waves; tile 128(M) x 64(N); BK = 32
__global__ __launch_bounds__(256) void gemm_bf16_wmma(
    const bf16_t* __restrict__ A, const bf16_t* __restrict__ Wt,
    const float* __restrict__ bias, const float* __restrict__ sbias,
    bf16_t* __restrict__ outb, float* __restrict__ outf,
    int M, int N, int K, int act) {
  __shared__ bf16_t ldsA[2][128 * 40];  // row stride 40 elems = 80B (16B aligned)
  __shared__ bf16_t ldsB[2][64 * 40];

  const int tid  = threadIdx.x;
  const int lane = tid & 31;
  const int wave = tid >> 5;
  const int wx = wave & 1, wy = wave >> 1;      // 2 x 4 wave grid
  const int row0 = blockIdx.y * 128;
  const int col0 = blockIdx.x * 64;
  const int lrow = lane & 15;
  const int hi   = lane >> 4;                   // 0 or 1
  const int kA   = hi * 8;                      // A: K-half base (elems)
  const int kB   = hi * 8;                      // B: K-half base (elems)

  // per-thread staging assignment: A rows tid/4 and tid/4+64, chunk (tid&3)*8; B row tid/4
  const int sr = tid >> 2;
  const int sc = (tid & 3) * 8;

  v8f acc[2][2];
  acc[0][0] = (v8f)0.f; acc[0][1] = (v8f)0.f;
  acc[1][0] = (v8f)0.f; acc[1][1] = (v8f)0.f;

  const int nk = K >> 5;

  auto stage = [&](int buf, int k0) {
#if HAS_ASYNC_LDS
    g2l_async16(&ldsA[buf][sr * 40 + sc],        A  + (long)(row0 + sr)      * K + k0 + sc);
    g2l_async16(&ldsA[buf][(sr + 64) * 40 + sc], A  + (long)(row0 + sr + 64) * K + k0 + sc);
    g2l_async16(&ldsB[buf][sr * 40 + sc],        Wt + (long)(col0 + sr)      * K + k0 + sc);
#else
    float4 t0 = *(const float4*)(A  + (long)(row0 + sr)      * K + k0 + sc);
    float4 t1 = *(const float4*)(A  + (long)(row0 + sr + 64) * K + k0 + sc);
    float4 t2 = *(const float4*)(Wt + (long)(col0 + sr)      * K + k0 + sc);
    *(float4*)&ldsA[buf][sr * 40 + sc]        = t0;
    *(float4*)&ldsA[buf][(sr + 64) * 40 + sc] = t1;
    *(float4*)&ldsB[buf][sr * 40 + sc]        = t2;
    if (k0 + 32 < K) __builtin_prefetch(A + (long)(row0 + sr) * K + k0 + 32, 0, 3);
#endif
  };

  stage(0, 0);
  wait_async_lds();
  __syncthreads();

  int cur = 0;
  for (int kt = 0; kt < nk; ++kt) {
    if (kt + 1 < nk) stage(cur ^ 1, (kt + 1) << 5);

    Frag a0, a1, b0, b1;
    const bf16_t* lA = ldsA[cur];
    const bf16_t* lB = ldsB[cur];
    const int ra = (wy * 32 + lrow) * 40;
    a0.h[0] = *(const v8bf*)&lA[ra + kA];
    a0.h[1] = *(const v8bf*)&lA[ra + kA + 16];
    a1.h[0] = *(const v8bf*)&lA[ra + 16 * 40 + kA];
    a1.h[1] = *(const v8bf*)&lA[ra + 16 * 40 + kA + 16];
    const int rb = (wx * 32 + lrow) * 40;
    b0.h[0] = *(const v8bf*)&lB[rb + kB];
    b0.h[1] = *(const v8bf*)&lB[rb + kB + 16];
    b1.h[0] = *(const v8bf*)&lB[rb + 16 * 40 + kB];
    b1.h[1] = *(const v8bf*)&lB[rb + 16 * 40 + kB + 16];

    acc[0][0] = __builtin_amdgcn_wmma_f32_16x16x32_bf16(false, a0.v, false, b0.v, (short)0, acc[0][0], false, false);
    acc[0][1] = __builtin_amdgcn_wmma_f32_16x16x32_bf16(false, a0.v, false, b1.v, (short)0, acc[0][1], false, false);
    acc[1][0] = __builtin_amdgcn_wmma_f32_16x16x32_bf16(false, a1.v, false, b0.v, (short)0, acc[1][0], false, false);
    acc[1][1] = __builtin_amdgcn_wmma_f32_16x16x32_bf16(false, a1.v, false, b1.v, (short)0, acc[1][1], false, false);

    wait_async_lds();
    __syncthreads();
    cur ^= 1;
  }

  const float sb = sbias ? *sbias : 0.f;
  #pragma unroll
  for (int mt = 0; mt < 2; ++mt)
    #pragma unroll
    for (int nt = 0; nt < 2; ++nt) {
      const int col = col0 + wx * 32 + nt * 16 + lrow;
      const float bv = (bias ? bias[col] : 0.f) + sb;
      #pragma unroll
      for (int i = 0; i < 8; ++i) {
        const int row = row0 + wy * 32 + mt * 16 + hi * 8 + i;
        float v = acc[mt][nt][i] + bv;
        if (act) v = leaky_f(v);
        const long o = (long)row * N + col;
        if (outb) outb[o] = (bf16_t)v;
        if (outf) outf[o] = v;
      }
    }
}

// ---------------- batched block-diagonal (additive) WMMA GEMM ----------------
// out[:, r*64 : r*64+64] = leaky( A[:, r*KIN : r*KIN+KIN] @ Wt[r]^T + bias[r] + *sbias )
template <int KIN>
__global__ __launch_bounds__(256) void additive_bf16_wmma(
    const bf16_t* __restrict__ A, int ldA,
    const bf16_t* __restrict__ Wt,    // [R][64][KIN]
    const float* __restrict__ bias,   // [R][64]
    const float* __restrict__ sbias,
    bf16_t* __restrict__ out, int ldOut) {
  constexpr int KPAD = (KIN < 32) ? 32 : KIN;
  constexpr int LSTR = 72;            // 144B row stride, 16B aligned
  constexpr int CPR  = KPAD / 8;      // 16B chunks per row
  __shared__ bf16_t ldsA[128 * LSTR];
  __shared__ bf16_t ldsB[64 * LSTR];

  const int tid  = threadIdx.x;
  const int lane = tid & 31;
  const int wave = tid >> 5;
  const int wx = wave & 1, wy = wave >> 1;
  const int r    = blockIdx.x;        // slot
  const int row0 = blockIdx.y * 128;
  const int lrow = lane & 15;
  const int hi   = lane >> 4;
  const int kA   = hi * 8;
  const int kB   = hi * 8;

  const bf16_t* Ab = A + (long)r * KIN;
  const bf16_t* Wb = Wt + (long)r * 64 * KIN;

  for (int i = tid; i < 128 * CPR; i += 256) {
    int rr = i / CPR, ch = i % CPR;
    if (ch * 8 < KIN)
      g2l_async16(&ldsA[rr * LSTR + ch * 8], Ab + (long)(row0 + rr) * ldA + ch * 8);
    else
      *(float4*)&ldsA[rr * LSTR + ch * 8] = make_float4(0.f, 0.f, 0.f, 0.f);
  }
  for (int i = tid; i < 64 * CPR; i += 256) {
    int rr = i / CPR, ch = i % CPR;
    if (ch * 8 < KIN)
      g2l_async16(&ldsB[rr * LSTR + ch * 8], Wb + (long)rr * KIN + ch * 8);
    else
      *(float4*)&ldsB[rr * LSTR + ch * 8] = make_float4(0.f, 0.f, 0.f, 0.f);
  }
  wait_async_lds();
  __syncthreads();

  v8f acc[2][2];
  acc[0][0] = (v8f)0.f; acc[0][1] = (v8f)0.f;
  acc[1][0] = (v8f)0.f; acc[1][1] = (v8f)0.f;

  #pragma unroll
  for (int kt = 0; kt < KPAD / 32; ++kt) {
    const int k0 = kt * 32;
    Frag a0, a1, b0, b1;
    const int ra = (wy * 32 + lrow) * LSTR + k0;
    a0.h[0] = *(const v8bf*)&ldsA[ra + kA];
    a0.h[1] = *(const v8bf*)&ldsA[ra + kA + 16];
    a1.h[0] = *(const v8bf*)&ldsA[ra + 16 * LSTR + kA];
    a1.h[1] = *(const v8bf*)&ldsA[ra + 16 * LSTR + kA + 16];
    const int rb = (wx * 32 + lrow) * LSTR + k0;
    b0.h[0] = *(const v8bf*)&ldsB[rb + kB];
    b0.h[1] = *(const v8bf*)&ldsB[rb + kB + 16];
    b1.h[0] = *(const v8bf*)&ldsB[rb + 16 * LSTR + kB];
    b1.h[1] = *(const v8bf*)&ldsB[rb + 16 * LSTR + kB + 16];

    acc[0][0] = __builtin_amdgcn_wmma_f32_16x16x32_bf16(false, a0.v, false, b0.v, (short)0, acc[0][0], false, false);
    acc[0][1] = __builtin_amdgcn_wmma_f32_16x16x32_bf16(false, a0.v, false, b1.v, (short)0, acc[0][1], false, false);
    acc[1][0] = __builtin_amdgcn_wmma_f32_16x16x32_bf16(false, a1.v, false, b0.v, (short)0, acc[1][0], false, false);
    acc[1][1] = __builtin_amdgcn_wmma_f32_16x16x32_bf16(false, a1.v, false, b1.v, (short)0, acc[1][1], false, false);
  }

  const float sb = sbias ? *sbias : 0.f;
  #pragma unroll
  for (int mt = 0; mt < 2; ++mt)
    #pragma unroll
    for (int nt = 0; nt < 2; ++nt) {
      const int colLocal = wx * 32 + nt * 16 + lrow;     // 0..63
      const float bv = bias[r * 64 + colLocal] + sb;
      #pragma unroll
      for (int i = 0; i < 8; ++i) {
        const int row = row0 + wy * 32 + mt * 16 + hi * 8 + i;
        float v = acc[mt][nt][i] + bv;
        v = leaky_f(v);
        out[(long)row * ldOut + r * 64 + colLocal] = (bf16_t)v;
      }
    }
}

// ---------------- final tiny layer: out[b] = t2[b,:]·W_last + b_last ----------------
__global__ __launch_bounds__(256) void last_dot(const float* __restrict__ t2,
                                                const float* __restrict__ Wl,
                                                const float* __restrict__ bl,
                                                float* __restrict__ out, int Mtot) {
  int i = blockIdx.x * 256 + threadIdx.x;
  if (i >= Mtot) return;
  float s = bl[0];
  const float* row = t2 + (long)i * 64;
  #pragma unroll
  for (int k = 0; k < 64; ++k) s += row[k] * Wl[k];
  out[i] = s;
}

extern "C" void kernel_launch(void* const* d_in, const int* in_sizes, int n_in,
                              void* d_out, int out_size, void* d_ws, size_t ws_size,
                              hipStream_t stream) {
  const float* x         = (const float*)d_in[0];
  const float* W_pca     = (const float*)d_in[1];
  const float* b_pca     = (const float*)d_in[2];
  const float* W1        = (const float*)d_in[3];
  const float* b1        = (const float*)d_in[4];
  const float* W2        = (const float*)d_in[5];
  const float* b2        = (const float*)d_in[6];
  const float* add1_W    = (const float*)d_in[7];
  const float* add1_b    = (const float*)d_in[8];
  const float* add1_bias = (const float*)d_in[9];
  const float* add2_W    = (const float*)d_in[10];
  const float* add2_b    = (const float*)d_in[11];
  const float* add2_bias = (const float*)d_in[12];
  const float* W_pca2    = (const float*)d_in[13];
  const float* b_pca2    = (const float*)d_in[14];
  const float* W_last    = (const float*)d_in[15];
  const float* b_last    = (const float*)d_in[16];
  float* out = (float*)d_out;

  const int B = 4096, P = 1024, R = 64, WD = 1024, AW = 64, KW = 16;

  char* ws = (char*)d_ws;
  bf16_t* act0 = (bf16_t*)ws;                           // 16M elems (32MB): ping
  bf16_t* act1 = (bf16_t*)(ws + ((size_t)32 << 20));    // 16M elems (32MB): pong
  bf16_t* wp   = (bf16_t*)(ws + ((size_t)64 << 20));    // weight area
  bf16_t* wpca_t  = wp; wp += (size_t)R * P;            // [64][1024]
  bf16_t* w1_t    = wp; wp += (size_t)WD * R;           // [1024][64]
  bf16_t* w2_t    = wp; wp += (size_t)WD * WD;          // [1024][1024]
  bf16_t* a1w_t   = wp; wp += (size_t)R * AW * KW;      // [64][64][16]
  bf16_t* a2w_t   = wp; wp += (size_t)R * AW * AW;      // [64][64][64]
  bf16_t* wpca2_t = wp; wp += (size_t)R * (R * AW);     // [64][4096]

  auto blocks = [](long n) { return dim3((unsigned)((n + 255) / 256)); };

  // ---- convert inputs / transpose-convert weights to bf16 ----
  cvt_bf16<<<blocks((long)B * P), 256, 0, stream>>>(x, act0, (long)B * P);
  cvt_t_bf16<<<blocks((long)P * R), 256, 0, stream>>>(W_pca, wpca_t, P, R, (long)P * R);
  cvt_t_bf16<<<blocks((long)R * WD), 256, 0, stream>>>(W1, w1_t, R, WD, (long)R * WD);
  cvt_t_bf16<<<blocks((long)WD * WD), 256, 0, stream>>>(W2, w2_t, WD, WD, (long)WD * WD);
  cvt_t_bf16<<<blocks((long)R * KW * AW), 256, 0, stream>>>(add1_W, a1w_t, KW, AW, (long)R * KW * AW);
  cvt_t_bf16<<<blocks((long)R * AW * AW), 256, 0, stream>>>(add2_W, a2w_t, AW, AW, (long)R * AW * AW);
  cvt_t_bf16<<<blocks((long)(R * AW) * R), 256, 0, stream>>>(W_pca2, wpca2_t, R * AW, R, (long)(R * AW) * R);

  // ---- t = x @ W_pca + b_pca  ->  act1 [4096][64] ----
  gemm_bf16_wmma<<<dim3(R / 64, B / 128), 256, 0, stream>>>(
      act0, wpca_t, b_pca, nullptr, act1, nullptr, B, R, P, 0);
  // ---- h1 = leaky(t @ W1 + b1) -> act0 [4096][1024] ----
  gemm_bf16_wmma<<<dim3(WD / 64, B / 128), 256, 0, stream>>>(
      act1, w1_t, b1, nullptr, act0, nullptr, B, WD, R, 1);
  // ---- h2 = leaky(h1 @ W2 + b2) -> act1 [4096][1024] ----
  gemm_bf16_wmma<<<dim3(WD / 64, B / 128), 256, 0, stream>>>(
      act0, w2_t, b2, nullptr, act1, nullptr, B, WD, WD, 1);
  // ---- additive layer 1 (block-diag, K=16) -> act0 [4096][4096] ----
  additive_bf16_wmma<16><<<dim3(R, B / 128), 256, 0, stream>>>(
      act1, WD, a1w_t, add1_b, add1_bias, act0, R * AW);
  // ---- additive layer 2 (block-diag, K=64) -> act1 [4096][4096] ----
  additive_bf16_wmma<64><<<dim3(R, B / 128), 256, 0, stream>>>(
      act0, R * AW, a2w_t, add2_b, add2_bias, act1, R * AW);
  // ---- t2 = a2 @ W_pca2 + b_pca2 -> f32 (reuse act0 region) ----
  float* t2f = (float*)act0;
  gemm_bf16_wmma<<<dim3(R / 64, B / 128), 256, 0, stream>>>(
      act1, wpca2_t, b_pca2, nullptr, nullptr, t2f, B, R, R * AW, 0);
  // ---- out = t2 @ W_last + b_last ----
  last_dot<<<blocks(B), 256, 0, stream>>>(t2f, W_last, b_last, out, B);
}
